// FTNeuralCDEEncoder_36129264894691
// MI455X (gfx1250) — matrix operand order repaired and checked by reading, not compiled, not run on hardware
//
#include <hip/hip_runtime.h>
#include <hip/hip_bf16.h>

// ---------------------------------------------------------------------------
// FT Neural CDE encoder for MI455X (gfx1250), wave32 + WMMA bf16.
//   B=32, L=256, FT=6 (C=7), H=512, INIT=1024, H*C=3584, evals = 255*4 = 1020
//
// Cost model (no runtime available): the RK4 chain is 1020 strictly
// sequential evals; each eval = 16x3584x512 GEMM (WMMA bf16) + LN(3584) +
// 3584*rows exact-erf GELUs + C=7 contraction. The erf contraction is the
// VALU bottleneck, so batch rows (independent!) are spread over 8 sync-free
// persistent workgroups (ROWS=4 each, M=16 WMMA tile zero-padded).
// W2 pre-packed to bf16 WMMA B-fragment layout = 3.67 MB, L2-resident
// (8 WGs x 1020 x 3.67 MB ~= 30 GB of L2 reads, ~7 MB from HBM once).
// A-fragments are register-resident across the 224-tile N sweep.
// ---------------------------------------------------------------------------

#define Bsz   32
#define Lsz   256
#define Hsz   512
#define Csz   7
#define HC    3584          // H * C
#define INITD 1024
#define ROWS  4             // batch rows per workgroup (8 WGs)
#define NT    224           // HC / 16  n-tiles
#define KBL   16            // Hsz / 32 k-blocks
#define NEVAL 1020          // 255 steps * 4 RK4 evals

typedef unsigned short u16;
typedef __attribute__((ext_vector_type(16))) __bf16 v16bf;
typedef __attribute__((ext_vector_type(8)))  float  v8f;

union Frag  { v16bf v; uint4 q[2]; };
union Pack8 { uint4 q; u16 h[8]; };

__device__ __forceinline__ u16 f2bf(float f) {
    unsigned u = __float_as_uint(f);
    u += 0x7fffu + ((u >> 16) & 1u);          // round-to-nearest-even
    return (u16)(u >> 16);
}
__device__ __forceinline__ float bf2f(u16 b) {
    return __uint_as_float(((unsigned)b) << 16);
}
__device__ __forceinline__ float gelu_erf(float x) {
    return 0.5f * x * (1.0f + erff(x * 0.70710678118654752f));
}

// ---------------------------------------------------------------------------
// Pack W2 (3584 x 512, row-major, f32) into bf16 WMMA B-fragment tiles.
// chunk = (nt*16 + kb)*32 + lane ; each lane owns 32 contiguous bytes:
//   n = nt*16 + (lane&15), k = kb*32 + (lane>>4)*16 .. +15 (pairs low=even K)
// ---------------------------------------------------------------------------
__global__ __launch_bounds__(256) void pack_w2(const float* __restrict__ W2,
                                               u16* __restrict__ W2p) {
    int d = blockIdx.x * 256 + threadIdx.x;      // dword id
    if (d >= HC * Hsz / 2) return;
    int j    = d & 7;
    int lane = (d >> 3) & 31;
    int kb   = (d >> 8) & 15;
    int nt   = d >> 12;
    int n = nt * 16 + (lane & 15);
    int k = kb * 32 + (lane >> 4) * 16 + 2 * j;
    unsigned lo = f2bf(W2[n * Hsz + k]);
    unsigned hi = f2bf(W2[n * Hsz + k + 1]);
    ((unsigned*)W2p)[d] = lo | (hi << 16);
}

// ---------------------------------------------------------------------------
// z0 = GELU(LN(img_feat @ Wi.T + bi)); one block per batch row (one-time cost)
// Also writes out[:,0,:] (the z0 endpoint of the returned state pair).
// ---------------------------------------------------------------------------
__global__ __launch_bounds__(256) void init_enc(const float* __restrict__ img,
                                                const float* __restrict__ Wi,
                                                const float* __restrict__ bi,
                                                const float* __restrict__ gi,
                                                const float* __restrict__ bei,
                                                float* __restrict__ z0,
                                                float* __restrict__ out) {
    __shared__ float sImg[INITD];
    __shared__ float sP[2 * 256];
    int b = blockIdx.x, tid = threadIdx.x;
    for (int q = tid; q < INITD; q += 256) sImg[q] = img[b * INITD + q];
    __syncthreads();
    float y[2];
    #pragma unroll
    for (int i = 0; i < 2; ++i) {
        int h = tid + i * 256;
        const float* w = Wi + (size_t)h * INITD;
        float acc = 0.f;
        for (int k = 0; k < INITD; k += 4) {
            acc = fmaf(sImg[k + 0], w[k + 0], acc);
            acc = fmaf(sImg[k + 1], w[k + 1], acc);
            acc = fmaf(sImg[k + 2], w[k + 2], acc);
            acc = fmaf(sImg[k + 3], w[k + 3], acc);
        }
        y[i] = acc + bi[h];
    }
    sP[tid] = y[0] + y[1];
    sP[256 + tid] = y[0] * y[0] + y[1] * y[1];
    __syncthreads();
    for (int off = 128; off > 0; off >>= 1) {
        if (tid < off) { sP[tid] += sP[tid + off]; sP[256 + tid] += sP[256 + tid + off]; }
        __syncthreads();
    }
    float mean = sP[0] * (1.0f / Hsz);
    float var  = sP[256] * (1.0f / Hsz) - mean * mean;
    float rstd = rsqrtf(var + 1e-5f);
    #pragma unroll
    for (int i = 0; i < 2; ++i) {
        int h = tid + i * 256;
        float o = gelu_erf((y[i] - mean) * rstd * gi[h] + bei[h]);
        z0[b * Hsz + h] = o;
        out[b * 2 * Hsz + h] = o;       // out[b][0][h]
    }
}

// ---------------------------------------------------------------------------
// te(t) = GELU(LN(t*Wt + bt)) for all 1020 eval times. One block per eval.
// ---------------------------------------------------------------------------
__global__ __launch_bounds__(256) void time_proj(const float* __restrict__ Wt,
                                                 const float* __restrict__ bt,
                                                 const float* __restrict__ gt,
                                                 const float* __restrict__ bet,
                                                 float* __restrict__ teTab) {
    __shared__ float sP[2 * 256];
    int ev = blockIdx.x, tid = threadIdx.x;
    int i = ev >> 2, e = ev & 3;
    float tval = (float)i + (float)e * (1.0f / 3.0f);
    float y[2];
    #pragma unroll
    for (int ii = 0; ii < 2; ++ii) {
        int h = tid + ii * 256;
        y[ii] = tval * Wt[h] + bt[h];
    }
    sP[tid] = y[0] + y[1];
    sP[256 + tid] = y[0] * y[0] + y[1] * y[1];
    __syncthreads();
    for (int off = 128; off > 0; off >>= 1) {
        if (tid < off) { sP[tid] += sP[tid + off]; sP[256 + tid] += sP[256 + tid + off]; }
        __syncthreads();
    }
    float mean = sP[0] * (1.0f / Hsz);
    float var  = sP[256] * (1.0f / Hsz) - mean * mean;
    float rstd = rsqrtf(var + 1e-5f);
    #pragma unroll
    for (int ii = 0; ii < 2; ++ii) {
        int h = tid + ii * 256;
        teTab[ev * Hsz + h] = gelu_erf((y[ii] - mean) * rstd * gt[h] + bet[h]);
    }
}

// ---------------------------------------------------------------------------
// dX table: Hermite cubic derivative of the control path at each eval time.
// dXTab[(ev*32 + b)*7 + c]
// ---------------------------------------------------------------------------
__global__ __launch_bounds__(256) void spline_dx(const float* __restrict__ ft,
                                                 float* __restrict__ dXTab) {
    int id = blockIdx.x * 256 + threadIdx.x;
    if (id >= NEVAL * Bsz * Csz) return;
    int c  = id % Csz;
    int r  = id / Csz;
    int b  = r & 31;
    int ev = r >> 5;
    int i = ev >> 2, e = ev & 3;
    float t   = (float)i + (float)e * (1.0f / 3.0f);
    int   idx = (e == 3) ? ((i + 1 < Lsz - 2) ? (i + 1) : (Lsz - 2)) : i;
    float s   = t - (float)idx;
    // x[b][l][0] = l/255 ; x[b][l][c] = ft[b][l][c-1]
    float x_im1 = (idx == 0) ? 0.f
                : (c == 0 ? (float)(idx - 1) * (1.0f / 255.0f)
                          : ft[(b * Lsz + idx - 1) * 6 + (c - 1)]);
    float x_i   = (c == 0) ? (float)idx * (1.0f / 255.0f)
                           : ft[(b * Lsz + idx) * 6 + (c - 1)];
    float x_ip1 = (c == 0) ? (float)(idx + 1) * (1.0f / 255.0f)
                           : ft[(b * Lsz + idx + 1) * 6 + (c - 1)];
    float mj = x_ip1 - x_i;
    float mp = (idx == 0) ? mj : (x_i - x_im1);
    float cbv = mp, ccv = 2.f * (mj - mp), cdv = mp - mj;
    dXTab[id] = cbv + s * (2.f * ccv + 3.f * s * cdv);
}

// ---------------------------------------------------------------------------
// Persistent CDE integrator: 8 workgroups x 256 threads (8 waves), 4 rows ea.
// A is a 16-row WMMA tile with rows 4..15 zero-padded. A-fragments are held
// in 128 VGPRs across the full 224-tile N sweep (no redundant LDS reloads).
// ---------------------------------------------------------------------------
__global__ __launch_bounds__(256, 1)
void cde_main(const u16* __restrict__ W2p, const float* __restrict__ b2,
              const float* __restrict__ g2, const float* __restrict__ be2,
              const float* __restrict__ teTab, const float* __restrict__ dXTab,
              const float* __restrict__ z0, float* __restrict__ out) {
    __shared__ alignas(16) u16 sY[ROWS * HC];     // 28672 B post-GEMM activ (bf16)
    __shared__ alignas(16) u16 sX[16 * Hsz];      // 16384 B WMMA A tile (padded M=16)
    __shared__ float sZ[ROWS * Hsz];              //  8192 B each
    __shared__ float sAcc[ROWS * Hsz];
    __shared__ float sK1[ROWS * Hsz];
    __shared__ float sK2[ROWS * Hsz];
    __shared__ float sArg[ROWS * Hsz];
    __shared__ float sStatP[2 * 256];
    __shared__ float sMean[ROWS], sRstd[ROWS];
    __shared__ float sDX[ROWS * Csz];

    const int tid = threadIdx.x;
    const int rowBase = blockIdx.x * ROWS;
    const int wave = tid >> 5, lane = tid & 31;
    const int mrow = lane & 15, part = lane >> 4;

    // init: z state + zero the padded A rows once (rows >= ROWS stay zero)
    for (int q = tid; q < 16 * Hsz; q += 256) sX[q] = 0;
    for (int q = tid; q < ROWS * Hsz; q += 256) {
        float v = z0[rowBase * Hsz + q];
        sZ[q] = v; sArg[q] = v;
    }
    __syncthreads();

    for (int ev = 0; ev < NEVAL; ++ev) {
        const int e = ev & 3;
        // -------- build A rows 0..ROWS-1 = bf16(arg + te(t)), fetch dX -----
        const float* te = teTab + ev * Hsz;
        for (int q = tid; q < ROWS * Hsz; q += 256) {
            int h = q & (Hsz - 1);
            sX[q] = f2bf(sArg[q] + te[h]);
        }
        if (tid < ROWS * Csz) {
            int m = tid / Csz, c = tid - m * Csz;
            sDX[tid] = dXTab[(ev * Bsz + rowBase + m) * Csz + c];
        }
        __syncthreads();

        // -------- GEMM: y = A @ W2.T (16 x 3584, K=512), bf16 WMMA ---------
        Frag aF[KBL];                      // register-resident A fragments
        #pragma unroll
        for (int kb = 0; kb < KBL; ++kb) {
            const uint4* ap = (const uint4*)(sX + mrow * Hsz + kb * 32 + part * 16);
            aF[kb].q[0] = ap[0];
            aF[kb].q[1] = ap[1];           // +8 bf16 elements
        }
        for (int nt = wave; nt < NT; nt += 8) {
            v8f acc = {};
            const uint4* bp = (const uint4*)W2p + ((size_t)(nt * KBL) * 32 + lane) * 2;
            #pragma unroll
            for (int kb = 0; kb < KBL; ++kb) {
                Frag bf;
                bf.q[0] = bp[0];
                bf.q[1] = bp[1];
                bp += 64;                  // next k-block: 32 lanes * 2 uint4
                acc = __builtin_amdgcn_wmma_f32_16x16x32_bf16(
                        false, aF[kb].v, false, bf.v, (short)0, acc, false, false);
            }
            // store only the ROWS valid rows of the C tile
            if (part == 0) {
                int col  = nt * 16 + mrow;
                float bias = b2[col];
                #pragma unroll
                for (int r = 0; r < ROWS; ++r)
                    sY[r * HC + col] = f2bf(acc[r] + bias);
            }
        }
        __syncthreads();

        // -------- LayerNorm stats over 3584 per row (vectorized reads) -----
        {
            int m = tid >> 6, j = tid & 63;              // 64 threads per row
            const uint4* yp = (const uint4*)(sY + m * HC + j * 56);
            float s = 0.f, ss = 0.f;
            #pragma unroll
            for (int w = 0; w < 7; ++w) {                // 7 x 8 bf16 = 56 elems
                Pack8 p; p.q = yp[w];
                #pragma unroll
                for (int x = 0; x < 8; ++x) {
                    float v = bf2f(p.h[x]); s += v; ss += v * v;
                }
            }
            sStatP[tid] = s; sStatP[256 + tid] = ss;
        }
        __syncthreads();
        if (tid < ROWS) {
            float s = 0.f, ss = 0.f;
            for (int j = 0; j < 64; ++j) {
                s  += sStatP[tid * 64 + j];
                ss += sStatP[256 + tid * 64 + j];
            }
            float mean = s * (1.0f / HC);
            float var  = ss * (1.0f / HC) - mean * mean;
            sMean[tid] = mean;
            sRstd[tid] = rsqrtf(var + 1e-5f);
        }
        __syncthreads();

        // -------- LN + GELU + contract with dX, RK4 (3/8 rule) combine -----
        for (int q = tid; q < ROWS * Hsz; q += 256) {
            int m = q >> 9, h = q & (Hsz - 1);
            float mean = sMean[m], rstd = sRstd[m];
            float kv = 0.f;
            #pragma unroll
            for (int c = 0; c < Csz; ++c) {
                int n = h * Csz + c;
                float o = (bf2f(sY[m * HC + n]) - mean) * rstd * g2[n] + be2[n];
                kv += gelu_erf(o) * sDX[m * Csz + c];
            }
            float na;
            if (e == 0)      { sK1[q] = kv; sAcc[q] = kv;
                               na = sZ[q] + kv * (1.0f / 3.0f); }
            else if (e == 1) { sK2[q] = kv; sAcc[q] += 3.0f * kv;
                               na = sZ[q] + kv - sK1[q] * (1.0f / 3.0f); }
            else if (e == 2) { sAcc[q] += 3.0f * kv;
                               na = sZ[q] + sK1[q] - sK2[q] + kv; }
            else             { float z = sZ[q] + (sAcc[q] + kv) * 0.125f;
                               sZ[q] = z; na = z; }
            sArg[q] = na;
        }
        __syncthreads();
    }

    for (int q = tid; q < ROWS * Hsz; q += 256) {
        int m = q >> 9, h = q & (Hsz - 1);
        out[(rowBase + m) * 2 * Hsz + Hsz + h] = sZ[q];   // out[b][1][h]
    }
}

// ---------------------------------------------------------------------------
extern "C" void kernel_launch(void* const* d_in, const int* in_sizes, int n_in,
                              void* d_out, int out_size, void* d_ws, size_t ws_size,
                              hipStream_t stream) {
    const float* img = (const float*)d_in[0];
    const float* ft  = (const float*)d_in[1];
    const float* Wi  = (const float*)d_in[2];
    const float* bi  = (const float*)d_in[3];
    const float* gi  = (const float*)d_in[4];
    const float* bei = (const float*)d_in[5];
    const float* Wt  = (const float*)d_in[6];
    const float* bt  = (const float*)d_in[7];
    const float* gt  = (const float*)d_in[8];
    const float* bet = (const float*)d_in[9];
    const float* W2  = (const float*)d_in[10];
    const float* b2  = (const float*)d_in[11];
    const float* g2  = (const float*)d_in[12];
    const float* be2 = (const float*)d_in[13];
    float* out = (float*)d_out;

    char* ws = (char*)d_ws;
    u16*   W2p   = (u16*)ws;                                   // 3,670,016 B
    float* teTab = (float*)(ws + 3670016);                     // 2,088,960 B
    float* dXTab = (float*)(ws + 3670016 + 2088960);           //   913,920 B
    float* z0    = (float*)(ws + 3670016 + 2088960 + 913920);  //    65,536 B

    pack_w2  <<<HC * Hsz / 2 / 256, 256, 0, stream>>>(W2, W2p);
    init_enc <<<Bsz, 256, 0, stream>>>(img, Wi, bi, gi, bei, z0, out);
    time_proj<<<NEVAL, 256, 0, stream>>>(Wt, bt, gt, bet, teTab);
    spline_dx<<<(NEVAL * Bsz * Csz + 255) / 256, 256, 0, stream>>>(ft, dXTab);
    cde_main <<<Bsz / ROWS, 256, 0, stream>>>(W2p, b2, g2, be2, teTab, dXTab, z0, out);
}
